// RPFusionPaper_58042188038462
// MI455X (gfx1250) — 8-bit matrix operand order repaired and checked by reading, compiled, not run on hardware
//
#include <hip/hip_runtime.h>
#include <stdint.h>


typedef __attribute__((ext_vector_type(8))) int   v8i;
typedef __attribute__((ext_vector_type(8))) float v8f;

#define BATCHES    32
#define C_OUT      64
#define T_DIM      256
#define HW         3136
#define NT         64
#define LDS_STRIDE 272
#define FP8_ONE    0x38u

__device__ __forceinline__ uint32_t pack4_fp8(uint32_t f0, uint32_t f1,
                                              uint32_t f2, uint32_t f3) {
    const uint32_t p01 = __builtin_amdgcn_perm(f1, f0, 0x00000703u);
    const uint32_t p23 = __builtin_amdgcn_perm(f3, f2, 0x00000703u);
    return __builtin_amdgcn_perm(p23, p01, 0x05040100u) & 0x38383838u;
}

__global__ void rp_build_afrag(const int* __restrict__ idx,
                               uint32_t* __restrict__ afrag) {
    const int tid   = threadIdx.x;
    const int w     = tid >> 8;
    const int s     = (tid >> 6) & 3;
    const int chunk = (tid >> 5) & 1;
    const int lane  = tid & 31;
    const int g     = lane >> 4;
    const int row   = w * 16 + (lane & 15);

    const int i0 = idx[row * 4 + 0];
    const int i1 = idx[row * 4 + 1];
    const int i2 = idx[row * 4 + 2];
    const int i3 = idx[row * 4 + 3];

    uint32_t d[4];
#pragma unroll
    for (int jj = 0; jj < 4; ++jj) {
        const int j = chunk * 4 + jj;
        uint32_t dw = 0u;
#pragma unroll
        for (int bb = 0; bb < 4; ++bb) {
            const int K = 32 * (j >> 2) + 16 * g + 4 * (j & 3) + bb;
            const int t = s * 64 + K;
            const uint32_t m =
                (t == i0 || t == i1 || t == i2 || t == i3) ? FP8_ONE : 0u;
            dw |= m << (8 * bb);
        }
        d[jj] = dw;
    }
    uint32_t* p = afrag + ((((w * 4 + s) * 2 + chunk) * 32 + lane) * 4);
    *reinterpret_cast<uint4*>(p) = make_uint4(d[0], d[1], d[2], d[3]);
}

__global__ void __launch_bounds__(128)
rp_wmma_gemm(const float* __restrict__ x,
             const uint32_t* __restrict__ afrag,
             float* __restrict__ out) {
    __shared__ alignas(16) unsigned char smem[NT * LDS_STRIDE];

    const int b   = blockIdx.y;
    const int hw0 = blockIdx.x * NT;
    const int tid = threadIdx.x;

    {
        const int col4 = (tid & 15) * 4;
        const float* base = x + (size_t)b * T_DIM * HW + hw0 + col4;
#pragma unroll 2
        for (int tg = (tid >> 4); tg < 64; tg += 8) {
            uint4 v[4];
#pragma unroll
            for (int tp = 0; tp < 4; ++tp)
                v[tp] = *reinterpret_cast<const uint4*>(
                    base + (size_t)(4 * tg + tp) * HW);
            const uint32_t u0 = pack4_fp8(v[0].x, v[1].x, v[2].x, v[3].x);
            const uint32_t u1 = pack4_fp8(v[0].y, v[1].y, v[2].y, v[3].y);
            const uint32_t u2 = pack4_fp8(v[0].z, v[1].z, v[2].z, v[3].z);
            const uint32_t u3 = pack4_fp8(v[0].w, v[1].w, v[2].w, v[3].w);
            const int kofs = 4 * tg;
            *reinterpret_cast<uint32_t*>(smem + (col4 + 0) * LDS_STRIDE + kofs) = u0;
            *reinterpret_cast<uint32_t*>(smem + (col4 + 1) * LDS_STRIDE + kofs) = u1;
            *reinterpret_cast<uint32_t*>(smem + (col4 + 2) * LDS_STRIDE + kofs) = u2;
            *reinterpret_cast<uint32_t*>(smem + (col4 + 3) * LDS_STRIDE + kofs) = u3;
        }
    }
    __syncthreads();

    const int wave = tid >> 5;
    const int lane = tid & 31;
    const int g    = lane >> 4;
    const int ln   = lane & 15;

    v8i a[4];
#pragma unroll
    for (int s = 0; s < 4; ++s) {
        const uint4 lo = *reinterpret_cast<const uint4*>(
            afrag + ((((wave * 4 + s) * 2 + 0) * 32 + lane) * 4));
        const uint4 hi = *reinterpret_cast<const uint4*>(
            afrag + ((((wave * 4 + s) * 2 + 1) * 32 + lane) * 4));
        a[s] = (v8i){(int)lo.x, (int)lo.y, (int)lo.z, (int)lo.w,
                     (int)hi.x, (int)hi.y, (int)hi.z, (int)hi.w};
    }

    v8f acc[4];
#pragma unroll
    for (int nt = 0; nt < 4; ++nt)
        acc[nt] = (v8f){0.f, 0.f, 0.f, 0.f, 0.f, 0.f, 0.f, 0.f};

#pragma unroll
    for (int s = 0; s < 4; ++s) {
        v8i bfr[4];
#pragma unroll
        for (int nt = 0; nt < 4; ++nt) {
            const unsigned char* bp =
                smem + (nt * 16 + ln) * LDS_STRIDE + s * 64 + g * 16;
            const uint4 lo = *reinterpret_cast<const uint4*>(bp);
            const uint4 hi = *reinterpret_cast<const uint4*>(bp + 32);
            bfr[nt] = (v8i){(int)lo.x, (int)lo.y, (int)lo.z, (int)lo.w,
                            (int)hi.x, (int)hi.y, (int)hi.z, (int)hi.w};
        }
#pragma unroll
        for (int nt = 0; nt < 4; ++nt)
            acc[nt] = __builtin_amdgcn_wmma_f32_16x16x64_fp8_fp8(
                a[s], bfr[nt], (short)0, acc[nt], false, false);
    }

#pragma unroll
    for (int nt = 0; nt < 4; ++nt) {
        const int colN = hw0 + nt * 16 + ln;
#pragma unroll
        for (int r = 0; r < 8; ++r) {
            const int   row = wave * 16 + g * 8 + r;
            const float sv  = acc[nt][r];
            out[(size_t)(b * C_OUT + row) * HW + colN] =
                (sv >= 2.0f) ? 1.0f : 0.0f;
        }
    }
}

extern "C" void kernel_launch(void* const* d_in, const int* in_sizes, int n_in,
                              void* d_out, int out_size, void* d_ws, size_t ws_size,
                              hipStream_t stream) {
    const float* x   = (const float*)d_in[0];
    const int*   idx = (const int*)d_in[1];
    float*       out = (float*)d_out;
    uint32_t*    afrag = (uint32_t*)d_ws;

    rp_build_afrag<<<dim3(1), dim3(1024), 0, stream>>>(idx, afrag);
    rp_wmma_gemm<<<dim3(HW / NT, BATCHES), dim3(128), 0, stream>>>(x, afrag, out);
}
